// FuturePredictionODE_7859790152323
// MI455X (gfx1250) — matrix operand order repaired and checked by reading, not compiled, not run on hardware
//
#include <hip/hip_runtime.h>
#include <hip/hip_bf16.h>

typedef unsigned short u16;
typedef __attribute__((ext_vector_type(8)))  float    v8f;
typedef __attribute__((ext_vector_type(16))) __bf16   v16bf;
typedef __attribute__((ext_vector_type(16))) u16      v16u;
typedef __attribute__((ext_vector_type(8)))  u16      v8u;
typedef __attribute__((ext_vector_type(4)))  unsigned v4ui;

constexpr int kW   = 128;
constexpr int kH   = 128;
constexpr int kHW  = kW * kH;          // 16384
constexpr int kC   = 64;
constexpr int kB   = 4;
constexpr int kFUT = 6;
constexpr int kSTEPS = 40;
constexpr int kCHW = kC * kHW;         // 1,048,576 floats per (b) state
constexpr int kSZ1 = kB * kCHW;        // 4,194,304 floats
constexpr float kDT = 0.05f;

__device__ __forceinline__ u16 f2bf(float f) {
    unsigned x = __builtin_bit_cast(unsigned, f);
    unsigned r = (x + 0x7FFFu + ((x >> 16) & 1u)) >> 16;
    return (u16)r;
}
__device__ __forceinline__ float sigm(float x) { return 1.f / (1.f + __expf(-x)); }

// ---------------------------------------------------------------------------
// Weight pack: OIHW f32 (cout,cin,3,3) -> bf16 [k = tap*cin+ci][cout]
// ---------------------------------------------------------------------------
__global__ void k_pack(const float* __restrict__ src, u16* __restrict__ dst,
                       int cin, int cout) {
    int total = cout * cin * 9;
    for (int i = blockIdx.x * blockDim.x + threadIdx.x; i < total;
         i += gridDim.x * blockDim.x) {
        int co = i / (cin * 9);
        int r  = i - co * (cin * 9);
        int ci = r / 9;
        int t  = r - ci * 9;
        dst[(t * cin + ci) * cout + co] = f2bf(src[i]);
    }
}

// ---------------------------------------------------------------------------
// Implicit-GEMM 3x3 conv via V_WMMA_F32_16X16X32_BF16 (templated on CIN).
//   Block tile: 64 pixels x 32 out-channels, 128 threads (4 waves).
//   Each wave: 16 px x 32 co = 2 WMMA accumulators per A fragment.
//   K = 9*CIN chunked by 32 (one tap x 32 input channels per chunk).
//   A panel double-buffered in LDS (one barrier/chunk); OOB handled by
//   clamped-pointer loads + bitmask zeroing (no per-element predication).
//   mode: 0 = store (bias), 1 = store relu, 2 = out += relu
// ---------------------------------------------------------------------------
template <int CIN>
__global__ __launch_bounds__(128)
void k_conv3x3(const float* __restrict__ X0, const float* __restrict__ X1,
               long long fs0, long long fs1,
               const u16* __restrict__ Wp, const float* __restrict__ bias,
               float* __restrict__ Out, long long fso,
               int cout, int dil, int mode) {
    constexpr int nK = (9 * CIN) / 32;
    constexpr int CSH = (CIN == 64) ? 1 : 2;      // kc>>CSH = tap
    constexpr int CMK = (CIN / 32) - 1;           // (kc&CMK)<<5 = ci0

    __shared__ u16 sA[2][64 * 32];                // double-buffered A panel
    __shared__ u16 sB[nK * 1024];                 // full 32-co weight panel

    const int tid  = threadIdx.x;
    const int lane = tid & 31;
    const int wv   = tid >> 5;
    const int f    = blockIdx.z;
    const int co0  = blockIdx.y << 5;             // 32 out channels per block
    const int pixBase = blockIdx.x << 6;          // 64 px, single image row
    const int y  = pixBase >> 7;
    const int xb = pixBase & (kW - 1);

    // ---- preload full weight panel: sB[kc][coj][kj] ----
    for (int i = tid; i < nK * 1024; i += 128) {
        int kc  = i >> 10;
        int r   = i & 1023;
        int kj  = r >> 5;
        int coj = r & 31;
        sB[(kc << 10) + (coj << 5) + kj] =
            Wp[(long long)((kc << 5) + kj) * cout + co0 + coj];
    }

    const float* Xf0 = X0 + (long long)f * fs0;
    const float* Xf1 = (CIN > 64) ? (X1 + (long long)f * fs1) : nullptr;

    const int pl  = tid & 63;                     // pixel within tile
    const int grp = tid >> 6;                     // 0..1 : K half

    // im2col loader: thread (pl,grp) writes K = grp*16 .. grp*16+15 for px pl
    auto loadA = [&](int kc, u16* dst) {
        const int tap = kc >> CSH;
        const int ci0 = (kc & CMK) << 5;
        const int t3  = tap / 3;
        const int ky = t3 - 1, kx = tap - t3 * 3 - 1;
        const int yin = y + ky * dil;
        const int xin = xb + pl + kx * dil;
        const bool ok = ((unsigned)yin < (unsigned)kH) &
                        ((unsigned)xin < (unsigned)kW);
        const float* base = (CIN == 64 || ci0 < 64)
            ? (Xf0 + (long long)ci0 * kHW)
            : (Xf1 + (long long)(ci0 - 64) * kHW);
        const float* chan = base + (long long)(grp << 4) * kHW;
        // clamp OOB threads to a guaranteed-valid address; mask result to 0
        const float* ps = ok ? (chan + yin * kW + xin) : chan;
        v8u t0, t1;
        #pragma unroll
        for (int e = 0; e < 8; ++e)
            t0[e] = f2bf(ps[(long long)e * kHW]);
        #pragma unroll
        for (int e = 0; e < 8; ++e)
            t1[e] = f2bf(ps[(long long)(e + 8) * kHW]);
        const unsigned m32 = ok ? 0xFFFFFFFFu : 0u;
        const v4ui m4 = {m32, m32, m32, m32};
        t0 = __builtin_bit_cast(v8u, __builtin_bit_cast(v4ui, t0) & m4);
        t1 = __builtin_bit_cast(v8u, __builtin_bit_cast(v4ui, t1) & m4);
        v8u* d = (v8u*)&dst[(pl << 5) + (grp << 4)];
        d[0] = t0;
        d[1] = t1;
    };

    v8f acc0 = {0.f, 0.f, 0.f, 0.f, 0.f, 0.f, 0.f, 0.f};
    v8f acc1 = acc0;

    loadA(0, sA[0]);
    __syncthreads();

    const int rowA = (wv << 4) + (lane & 15);
    const int oA   = (lane >= 16) ? 16 : 0;       // A: K0-15 / K16-31
    const int col  = lane & 15;
    const int oB   = (lane >= 16) ? 16 : 0;       // B: K0-15 / K16-31

    #pragma unroll 2
    for (int kc = 0; kc < nK; ++kc) {
        if (kc + 1 < nK) loadA(kc + 1, sA[(kc + 1) & 1]);
        // prefetch chunk kc+2 (global_prefetch_b8)
        if (kc + 2 < nK) {
            const int kc2 = kc + 2;
            const int tap2 = kc2 >> CSH;
            const int ci02 = (kc2 & CMK) << 5;
            const int t32 = tap2 / 3;
            const int yin2 = y + (t32 - 1) * dil;
            const int xin2 = xb + pl + (tap2 - t32 * 3 - 1) * dil;
            if (((unsigned)yin2 < (unsigned)kH) &
                ((unsigned)xin2 < (unsigned)kW)) {
                const float* b2 = (CIN == 64 || ci02 < 64)
                    ? (Xf0 + (long long)ci02 * kHW)
                    : (Xf1 + (long long)(ci02 - 64) * kHW);
                __builtin_prefetch(
                    b2 + (long long)(grp << 4) * kHW + yin2 * kW + xin2, 0, 1);
            }
        }

        const u16* buf = sA[kc & 1];
        const v8u a0 = *(const v8u*)&buf[(rowA << 5) + oA];
        const v8u a1 = *(const v8u*)&buf[(rowA << 5) + oA + 8];
        const v8u b0 = *(const v8u*)&sB[(kc << 10) + (col << 5) + oB];
        const v8u b1 = *(const v8u*)&sB[(kc << 10) + (col << 5) + oB + 8];
        const v8u c0 = *(const v8u*)&sB[(kc << 10) + ((col + 16) << 5) + oB];
        const v8u c1 = *(const v8u*)&sB[(kc << 10) + ((col + 16) << 5) + oB + 8];

        v16u au, bu, cu;
        #pragma unroll
        for (int j = 0; j < 8; ++j) {
            au[j] = a0[j]; au[8 + j] = a1[j];
            bu[j] = b0[j]; bu[8 + j] = b1[j];
            cu[j] = c0[j]; cu[8 + j] = c1[j];
        }
        const v16bf av = __builtin_bit_cast(v16bf, au);
        acc0 = __builtin_amdgcn_wmma_f32_16x16x32_bf16(
            false, av, false, __builtin_bit_cast(v16bf, bu),
            (short)0, acc0, false, false);
        acc1 = __builtin_amdgcn_wmma_f32_16x16x32_bf16(
            false, av, false, __builtin_bit_cast(v16bf, cu),
            (short)0, acc1, false, false);
        __syncthreads();
    }

    // Epilogue: lane -> (N=channel, half), acc[r] -> pixel
    const int n = lane & 15, half = lane >> 4;
    const long long pixOff = pixBase + (wv << 4) + (half << 3);
    #pragma unroll
    for (int t = 0; t < 2; ++t) {
        const int co = co0 + n + (t << 4);
        const float bv = bias[co];
        const v8f* A = t ? &acc1 : &acc0;
        float* op = Out + (long long)f * fso + (long long)co * kHW + pixOff;
        #pragma unroll
        for (int r = 0; r < 8; ++r) {
            float v = (*A)[r] + bv;
            if (mode == 1) v = fmaxf(v, 0.f);
            if (mode == 2) op[r] += fmaxf(v, 0.f);
            else           op[r] = v;
        }
    }
}

// ---------------------------------------------------------------------------
// Elementwise GRU kernels (fp32 VALU)
// ---------------------------------------------------------------------------
__global__ __launch_bounds__(256)
void k_sig_mul(float* __restrict__ out, const float* __restrict__ a,
               const float* __restrict__ b) {
    int i = blockIdx.x * 256 + threadIdx.x;
    out[i] = sigm(a[i]) * b[i];
}

__global__ __launch_bounds__(256)
void k_ode_upd(float* __restrict__ h, const float* __restrict__ z,
               const float* __restrict__ g) {
    int i = blockIdx.x * 256 + threadIdx.x;
    float hv = h[i];
    h[i] = hv + kDT * (1.f - sigm(z[i])) * (tanhf(g[i]) - hv);
}

__global__ __launch_bounds__(256)
void k_obs_gather(float* __restrict__ out, const float* __restrict__ cam,
                  const float* __restrict__ lid, const int* __restrict__ cts,
                  const int* __restrict__ lts, int t) {
    int i = blockIdx.x * 256 + threadIdx.x;
    int b = i / kCHW;
    int r = i - b * kCHW;
    float s = 0.f;
    #pragma unroll
    for (int n = 0; n < 3; ++n) {
        if (cts[b * 3 + n] == t) s += cam[(long long)(b * 3 + n) * kCHW + r];
        if (lts[b * 3 + n] == t) s += lid[(long long)(b * 3 + n) * kCHW + r];
    }
    out[i] = s;
}

__global__ __launch_bounds__(256)
void k_obs_comb(float* __restrict__ h, const float* __restrict__ z,
                const float* __restrict__ q, const int* __restrict__ cts,
                const int* __restrict__ lts, int t) {
    int i = blockIdx.x * 256 + threadIdx.x;
    int b = i / kCHW;
    bool has = false;
    #pragma unroll
    for (int n = 0; n < 3; ++n) {
        has |= (cts[b * 3 + n] == t);
        has |= (lts[b * 3 + n] == t);
    }
    float zz = sigm(z[i]);
    float hn = (1.f - zz) * tanhf(q[i]) + zz * h[i];
    if (has) h[i] = hn;
}

__global__ __launch_bounds__(256)
void k_pred_acc(float* __restrict__ out, const float* __restrict__ h,
                const int* __restrict__ tts, int t) {
    int fut = blockIdx.y, b = blockIdx.z;
    if (tts[b * kFUT + fut] != t) return;
    int i = blockIdx.x * 256 + threadIdx.x;
    out[(long long)(b * kFUT + fut) * kCHW + i] += h[(long long)b * kCHW + i];
}

__global__ __launch_bounds__(256)
void k_sgru_comb(float* __restrict__ hc, const float* __restrict__ z,
                 const float* __restrict__ q, float* __restrict__ xs) {
    int i = blockIdx.x * 256 + threadIdx.x;
    int b = i / kCHW;
    int r = i - b * kCHW;
    float zz = sigm(z[i]);
    float hn = (1.f - zz) * tanhf(q[i]) + zz * hc[i];
    hc[i] = hn;
    xs[(long long)b * (kFUT * kCHW) + r] = hn;   // x[:, s] in-place update
}

// ---------------------------------------------------------------------------
// Host orchestration
// ---------------------------------------------------------------------------
static inline void conv3x3(hipStream_t s, const float* X0, const float* X1,
                           long long fs0, long long fs1, const u16* Wp,
                           const float* bias, float* out, long long fso,
                           int F, int cin, int cout, int dil, int mode) {
    dim3 g(kHW / 64, cout / 32, F);
    if (cin == 64)
        k_conv3x3<64><<<g, 128, 0, s>>>(X0, X1, fs0, fs1, Wp, bias, out, fso,
                                        cout, dil, mode);
    else
        k_conv3x3<128><<<g, 128, 0, s>>>(X0, X1, fs0, fs1, Wp, bias, out, fso,
                                         cout, dil, mode);
}

extern "C" void kernel_launch(void* const* d_in, const int* in_sizes, int n_in,
                              void* d_out, int out_size, void* d_ws, size_t ws_size,
                              hipStream_t stream) {
    (void)in_sizes; (void)n_in; (void)ws_size;
    const float* fpi   = (const float*)d_in[0];
    const float* cam   = (const float*)d_in[1];
    const float* lid   = (const float*)d_in[2];
    const int*   cts   = (const int*)d_in[3];
    const int*   lts   = (const int*)d_in[4];
    const int*   tts   = (const int*)d_in[5];
    const float* odeWr = (const float*)d_in[6];  const float* ode_br = (const float*)d_in[7];
    const float* odeWz = (const float*)d_in[8];  const float* ode_bz = (const float*)d_in[9];
    const float* odeWg = (const float*)d_in[10]; const float* ode_bg = (const float*)d_in[11];
    const float* obsWz = (const float*)d_in[12]; const float* obs_bz = (const float*)d_in[13];
    const float* obsWr = (const float*)d_in[14]; const float* obs_br = (const float*)d_in[15];
    const float* obsWq = (const float*)d_in[16]; const float* obs_bq = (const float*)d_in[17];
    const float* s0Wz  = (const float*)d_in[18]; const float* s0_bz  = (const float*)d_in[19];
    const float* s0Wr  = (const float*)d_in[20]; const float* s0_br  = (const float*)d_in[21];
    const float* s0Wq  = (const float*)d_in[22]; const float* s0_bq  = (const float*)d_in[23];
    const float* s1Wz  = (const float*)d_in[24]; const float* s1_bz  = (const float*)d_in[25];
    const float* s1Wr  = (const float*)d_in[26]; const float* s1_br  = (const float*)d_in[27];
    const float* s1Wq  = (const float*)d_in[28]; const float* s1_bq  = (const float*)d_in[29];
    const float* blkW  = (const float*)d_in[30]; const float* blk_b  = (const float*)d_in[31];
    const float* d1W   = (const float*)d_in[32]; const float* d1_b   = (const float*)d_in[33];
    const float* d6W   = (const float*)d_in[34]; const float* d6_b   = (const float*)d_in[35];
    const float* d12W  = (const float*)d_in[36]; const float* d12_b  = (const float*)d_in[37];
    const float* prW   = (const float*)d_in[38]; const float* pr_b   = (const float*)d_in[39];

    float* F32 = (float*)d_ws;
    float* h      = F32;
    float* tmpA   = F32 + 1LL * kSZ1;
    float* tmpB   = F32 + 2LL * kSZ1;
    float* tmpC   = F32 + 3LL * kSZ1;
    float* rh     = F32 + 4LL * kSZ1;
    float* obs    = F32 + 5LL * kSZ1;
    float* hidden = F32 + 6LL * kSZ1;
    float* hc     = F32 + 7LL * kSZ1;
    float* xbuf   = F32 + 8LL * kSZ1;            // B*6*C*HW
    float* yfr    = F32 + 14LL * kSZ1;           // one 128-ch frame
    u16*   WP     = (u16*)(F32 + 15LL * kSZ1);

    const int K64  = 576 * 64;    // 64->64
    const int K128 = 1152 * 64;   // 128->64
    const int K64b = 576 * 128;   // 64->128
    u16* pOdeWr = WP;             u16* pOdeWz = pOdeWr + K64;
    u16* pOdeWg = pOdeWz + K64;   u16* pBlk   = pOdeWg + K64;
    u16* pObsWz = pBlk + K64;     u16* pObsWr = pObsWz + K128;
    u16* pObsWq = pObsWr + K128;  u16* pS0Wz  = pObsWq + K128;
    u16* pS0Wr  = pS0Wz + K128;   u16* pS0Wq  = pS0Wr + K128;
    u16* pS1Wz  = pS0Wq + K128;   u16* pS1Wr  = pS1Wz + K128;
    u16* pS1Wq  = pS1Wr + K128;   u16* pProj  = pS1Wq + K128;
    u16* pD1    = pProj + K128;   u16* pD6    = pD1 + K64b;
    u16* pD12   = pD6 + K64b;

    k_pack<<<128, 256, 0, stream>>>(odeWr, pOdeWr, 64, 64);
    k_pack<<<128, 256, 0, stream>>>(odeWz, pOdeWz, 64, 64);
    k_pack<<<128, 256, 0, stream>>>(odeWg, pOdeWg, 64, 64);
    k_pack<<<128, 256, 0, stream>>>(blkW,  pBlk,   64, 64);
    k_pack<<<128, 256, 0, stream>>>(obsWz, pObsWz, 128, 64);
    k_pack<<<128, 256, 0, stream>>>(obsWr, pObsWr, 128, 64);
    k_pack<<<128, 256, 0, stream>>>(obsWq, pObsWq, 128, 64);
    k_pack<<<128, 256, 0, stream>>>(s0Wz,  pS0Wz,  128, 64);
    k_pack<<<128, 256, 0, stream>>>(s0Wr,  pS0Wr,  128, 64);
    k_pack<<<128, 256, 0, stream>>>(s0Wq,  pS0Wq,  128, 64);
    k_pack<<<128, 256, 0, stream>>>(s1Wz,  pS1Wz,  128, 64);
    k_pack<<<128, 256, 0, stream>>>(s1Wr,  pS1Wr,  128, 64);
    k_pack<<<128, 256, 0, stream>>>(s1Wq,  pS1Wq,  128, 64);
    k_pack<<<128, 256, 0, stream>>>(prW,   pProj,  128, 64);
    k_pack<<<128, 256, 0, stream>>>(d1W,   pD1,    64, 128);
    k_pack<<<128, 256, 0, stream>>>(d6W,   pD6,    64, 128);
    k_pack<<<128, 256, 0, stream>>>(d12W,  pD12,   64, 128);

    hipMemsetAsync(d_out, 0, (size_t)out_size * sizeof(float), stream);
    hipMemcpyAsync(h, fpi, (size_t)kSZ1 * sizeof(float),
                   hipMemcpyDeviceToDevice, stream);

    const int ewB = kSZ1 / 256;
    // --------- ODE scan over 40 steps ---------
    for (int t = 0; t < kSTEPS; ++t) {
        conv3x3(stream, h, nullptr, kCHW, 0, pOdeWr, ode_br, tmpA, kCHW, kB, 64, 64, 1, 0);
        conv3x3(stream, h, nullptr, kCHW, 0, pOdeWz, ode_bz, tmpB, kCHW, kB, 64, 64, 1, 0);
        k_sig_mul<<<ewB, 256, 0, stream>>>(rh, tmpA, h);
        conv3x3(stream, rh, nullptr, kCHW, 0, pOdeWg, ode_bg, tmpC, kCHW, kB, 64, 64, 1, 0);
        k_ode_upd<<<ewB, 256, 0, stream>>>(h, tmpB, tmpC);
        k_obs_gather<<<ewB, 256, 0, stream>>>(obs, cam, lid, cts, lts, t);
        conv3x3(stream, obs, h, kCHW, kCHW, pObsWz, obs_bz, tmpA, kCHW, kB, 128, 64, 1, 0);
        conv3x3(stream, obs, h, kCHW, kCHW, pObsWr, obs_br, tmpB, kCHW, kB, 128, 64, 1, 0);
        k_sig_mul<<<ewB, 256, 0, stream>>>(rh, tmpB, h);
        conv3x3(stream, obs, rh, kCHW, kCHW, pObsWq, obs_bq, tmpC, kCHW, kB, 128, 64, 1, 0);
        k_obs_comb<<<ewB, 256, 0, stream>>>(h, tmpA, tmpC, cts, lts, t);
        k_pred_acc<<<dim3(kCHW / 256, kFUT, kB), 256, 0, stream>>>(
            (float*)d_out, h, tts, t);
    }

    // hidden = x[:,0]
    for (int b = 0; b < kB; ++b)
        hipMemcpyAsync(hidden + (long long)b * kCHW,
                       (float*)d_out + (long long)b * kFUT * kCHW,
                       (size_t)kCHW * sizeof(float), hipMemcpyDeviceToDevice, stream);

    // --------- spatial GRU layer 0 (x lives in d_out) ---------
    hipMemcpyAsync(hc, hidden, (size_t)kSZ1 * sizeof(float),
                   hipMemcpyDeviceToDevice, stream);
    for (int s = 0; s < kFUT; ++s) {
        float* xs = (float*)d_out + (long long)s * kCHW;  // b-stride = 6*kCHW
        conv3x3(stream, xs, hc, (long long)kFUT * kCHW, kCHW, pS0Wz, s0_bz, tmpA, kCHW, kB, 128, 64, 1, 0);
        conv3x3(stream, xs, hc, (long long)kFUT * kCHW, kCHW, pS0Wr, s0_br, tmpB, kCHW, kB, 128, 64, 1, 0);
        k_sig_mul<<<ewB, 256, 0, stream>>>(rh, tmpB, hc);
        conv3x3(stream, xs, rh, (long long)kFUT * kCHW, kCHW, pS0Wq, s0_bq, tmpC, kCHW, kB, 128, 64, 1, 0);
        k_sgru_comb<<<ewB, 256, 0, stream>>>(hc, tmpA, tmpC, xs);
    }
    conv3x3(stream, (float*)d_out, nullptr, kCHW, 0, pBlk, blk_b, xbuf, kCHW,
            kB * kFUT, 64, 64, 1, 1);

    // --------- spatial GRU layer 1 (x lives in xbuf) ---------
    hipMemcpyAsync(hc, hidden, (size_t)kSZ1 * sizeof(float),
                   hipMemcpyDeviceToDevice, stream);
    for (int s = 0; s < kFUT; ++s) {
        float* xs = xbuf + (long long)s * kCHW;
        conv3x3(stream, xs, hc, (long long)kFUT * kCHW, kCHW, pS1Wz, s1_bz, tmpA, kCHW, kB, 128, 64, 1, 0);
        conv3x3(stream, xs, hc, (long long)kFUT * kCHW, kCHW, pS1Wr, s1_br, tmpB, kCHW, kB, 128, 64, 1, 0);
        k_sig_mul<<<ewB, 256, 0, stream>>>(rh, tmpB, hc);
        conv3x3(stream, xs, rh, (long long)kFUT * kCHW, kCHW, pS1Wq, s1_bq, tmpC, kCHW, kB, 128, 64, 1, 0);
        k_sgru_comb<<<ewB, 256, 0, stream>>>(hc, tmpA, tmpC, xs);
    }

    // --------- ASPP head, per frame ---------
    for (int fr = 0; fr < kB * kFUT; ++fr) {
        const float* xin = xbuf + (long long)fr * kCHW;
        conv3x3(stream, xin, nullptr, 0, 0, pD1,  d1_b,  yfr, 0, 1, 64, 128, 1,  1);
        conv3x3(stream, xin, nullptr, 0, 0, pD6,  d6_b,  yfr, 0, 1, 64, 128, 6,  2);
        conv3x3(stream, xin, nullptr, 0, 0, pD12, d12_b, yfr, 0, 1, 64, 128, 12, 2);
        conv3x3(stream, yfr, yfr + 64LL * kHW, 0, 0, pProj, pr_b,
                (float*)d_out + (long long)fr * kCHW, 0, 1, 128, 64, 1, 0);
    }
}
